// biLstmCrf_19052474925737
// MI455X (gfx1250) — matrix operand order repaired and checked by reading, not compiled, not run on hardware
//
#include <hip/hip_runtime.h>

// Problem constants (from reference)
#define V_  50000
#define E_  512
#define H_  512
#define L_  128
#define B_  64
#define T_  512
#define G_  2048    // 4*H
#define BT_ 32768   // B*T

typedef __attribute__((ext_vector_type(16))) __bf16 v16bf;
typedef __attribute__((ext_vector_type(8)))  float  v8f;
typedef __attribute__((ext_vector_type(8)))  unsigned short ushort8;

union AFrag { v16bf v; ushort8 h[2]; };

__device__ inline unsigned short f2bf(float f) {
    unsigned int u = __float_as_uint(f);
    unsigned int r = u + 0x7FFFu + ((u >> 16) & 1u);   // round-to-nearest-even
    return (unsigned short)(r >> 16);
}
__device__ inline float sigmoidf_(float x) { return 1.0f / (1.0f + __expf(-x)); }

// Build an A fragment (16x32 bf16, ISA 7.12.2 layout) from a row-major
// bf16 buffer with row stride `stride` halves: lane holds row (lane&15),
// K-halves [hi*8 .. hi*8+7] and [hi*8+16 .. hi*8+23] of the 32-wide k-tile.
__device__ inline AFrag load_afrag(const unsigned short* rowbase) {
    AFrag a;
    a.h[0] = *(const ushort8*)(rowbase);
    a.h[1] = *(const ushort8*)(rowbase + 16);
    return a;
}

// ---------------------------------------------------------------------------
// Pack a row-major weight W[N][K] (g = h @ W.T) into WMMA B-fragment order:
// B[k][n] = W[n][k]; tile = (ntile, ktile) of 16 cols x 32 K; lane L holds
// col n = ntile*16 + (L&15), K = ktile*32 + (L>>4)*8 + (e&7) + (e>>3)*16 (table order).
// ---------------------------------------------------------------------------
__global__ void crf_pack_b_kernel(const float* __restrict__ W,
                                  unsigned short* __restrict__ out,
                                  int N, int K) {
    int tid = blockIdx.x * blockDim.x + threadIdx.x;
    if (tid >= N * K) return;
    const int KT = K >> 5;
    int e    = tid & 15;
    int lane = (tid >> 4) & 31;
    int rest = tid >> 9;
    int ktile = rest % KT;
    int ntile = rest / KT;
    int n = ntile * 16 + (lane & 15);
    int k = ktile * 32 + (lane >> 4) * 8 + (e & 7) + (e >> 3) * 16;
    out[tid] = f2bf(W[(size_t)n * K + k]);
}

// Embedding gather -> bf16, PAD(=0) token rows zeroed.
__global__ void crf_embed_kernel(const int* __restrict__ tokens,
                                 const float* __restrict__ table,
                                 unsigned short* __restrict__ x) {
    int tid = blockIdx.x * blockDim.x + threadIdx.x;   // < BT*E
    int bt = tid >> 9;
    int e  = tid & 511;
    int tok = tokens[bt];
    float v = (tok == 0) ? 0.0f : table[(size_t)tok * E_ + e];
    x[tid] = f2bf(v);
}

__global__ void crf_transexp_kernel(const float* __restrict__ tr,
                                    float* __restrict__ out) {
    int tid = blockIdx.x * blockDim.x + threadIdx.x;
    if (tid < L_ * L_) out[tid] = __expf(tr[tid]);
}

// ---------------------------------------------------------------------------
// gx = x @ w_ih.T + b  : (BT x 512) * (512 x 2048), bf16 WMMA, f32 out.
// A tile (16x512) staged once in LDS per block; B fragments double-buffered
// so the K loop overlaps L2 latency with WMMA issue.
// Block: 8 waves; wave w -> 4 N-tiles at nch*512 + w*64. Grid (BT/16, 4).
// ---------------------------------------------------------------------------
__global__ __launch_bounds__(256, 1) void crf_gemm_gx_kernel(
    const unsigned short* __restrict__ xbf,
    const unsigned short* __restrict__ pwih,
    const float* __restrict__ bias,
    float* __restrict__ gx) {
    __shared__ unsigned short sA[16 * 512];   // 16 KB
    const int lane = threadIdx.x & 31;
    const int wave = threadIdx.x >> 5;
    const int rt   = blockIdx.x;
    const int hi   = lane >> 4;
    const int nbase = blockIdx.y * 512 + wave * 64;

    // Stage A tile into LDS (coalesced 16B chunks).
    for (int i = threadIdx.x * 8; i < 16 * 512; i += 256 * 8) {
        *(ushort8*)&sA[i] =
            *(const ushort8*)&xbf[(size_t)(rt * 16 + (i >> 9)) * 512 + (i & 511)];
    }
    __syncthreads();

    v8f acc[4];
#pragma unroll
    for (int tt = 0; tt < 4; ++tt) {
        const float bv = bias[nbase + tt * 16 + (lane & 15)];
#pragma unroll
        for (int j = 0; j < 8; ++j) acc[tt][j] = bv;
    }

    const unsigned short* pB[4];
#pragma unroll
    for (int tt = 0; tt < 4; ++tt) {
        const int ntile = (nbase >> 4) + tt;
        pB[tt] = pwih + ((size_t)(ntile * 16) * 32 + lane) * 16;
    }
    const unsigned short* arow = &sA[(lane & 15) * 512 + hi * 8];

    v16bf bb[2][4];
#pragma unroll
    for (int tt = 0; tt < 4; ++tt) bb[0][tt] = *(const v16bf*)(pB[tt]);

    for (int kt = 0; kt < 16; ++kt) {
        const int cb = kt & 1, nb = cb ^ 1;
        if (kt + 1 < 16) {
#pragma unroll
            for (int tt = 0; tt < 4; ++tt)
                bb[nb][tt] = *(const v16bf*)(pB[tt] + (kt + 1) * 512);
        }
        AFrag a = load_afrag(arow + kt * 32);
#pragma unroll
        for (int tt = 0; tt < 4; ++tt)
            acc[tt] = __builtin_amdgcn_wmma_f32_16x16x32_bf16(
                false, a.v, false, bb[cb][tt], (short)0, acc[tt], false, false);
    }

#pragma unroll
    for (int tt = 0; tt < 4; ++tt) {
        const int n = nbase + tt * 16 + (lane & 15);
#pragma unroll
        for (int j = 0; j < 8; ++j) {
            const int m = rt * 16 + j + 8 * hi;
            gx[(size_t)m * G_ + n] = acc[tt][j];
        }
    }
}

// ---------------------------------------------------------------------------
// Persistent BiLSTM recurrence. Grid = 8 blocks (2 dirs x 4 batch-tiles of 16),
// 1024 threads = 32 waves. Per step: g = gx[t] + h @ w_hh.T via bf16 WMMA
// (wave w owns columns w*16.. in each of the 4 gate blocks -> gate math and
// the cell state c stay in accumulator VGPRs for all 512 steps).
// h ping-pongs through LDS (bf16); B fragments double-buffered over K.
// ---------------------------------------------------------------------------
__global__ __launch_bounds__(1024, 1) void crf_lstm_kernel(
    const float* __restrict__ gx_f, const float* __restrict__ gx_b,
    const unsigned short* __restrict__ pwhh_f,
    const unsigned short* __restrict__ pwhh_b,
    unsigned short* __restrict__ hcat) {
    __shared__ unsigned short hbuf[2][16 * 512];   // 32 KB
    const int dir  = blockIdx.x >> 2;
    const int b0   = (blockIdx.x & 3) * 16;
    const int lane = threadIdx.x & 31;
    const int wave = threadIdx.x >> 5;
    const float* gx = dir ? gx_b : gx_f;
    const unsigned short* pw = dir ? pwhh_b : pwhh_f;

    for (int i = threadIdx.x; i < 16 * 512; i += 1024) hbuf[0][i] = 0;
    __syncthreads();

    const int rowA = lane & 15;
    const int hi   = lane >> 4;
    const int hcol = wave * 16 + (lane & 15);   // hidden column 0..511

    const unsigned short* pB[4];
#pragma unroll
    for (int g = 0; g < 4; ++g) {
        const int ntile = g * 32 + wave;
        pB[g] = pw + ((size_t)(ntile * 16) * 32 + lane) * 16;
    }

    float cst[8];
#pragma unroll
    for (int j = 0; j < 8; ++j) cst[j] = 0.0f;

    int cur = 0;
    for (int s = 0; s < T_; ++s) {
        const int t = dir ? (T_ - 1 - s) : s;
        if (s + 1 < T_) {   // hint next step's gate-input rows into cache
            const int tn = dir ? (T_ - 2 - s) : (s + 1);
            __builtin_prefetch(&gx[(size_t)((b0 + rowA) * T_ + tn) * G_ + hcol], 0, 1);
        }
        v8f acc[4];
#pragma unroll
        for (int g = 0; g < 4; ++g) {
            const int col = g * 512 + hcol;
#pragma unroll
            for (int j = 0; j < 8; ++j) {
                const int m = j + 8 * hi;
                acc[g][j] = gx[(size_t)((b0 + m) * T_ + t) * G_ + col];
            }
        }
        v16bf bb[2][4];
#pragma unroll
        for (int g = 0; g < 4; ++g) bb[0][g] = *(const v16bf*)(pB[g]);

        for (int kt = 0; kt < 16; ++kt) {
            const int cb = kt & 1, nb = cb ^ 1;
            if (kt + 1 < 16) {
#pragma unroll
                for (int g = 0; g < 4; ++g)
                    bb[nb][g] = *(const v16bf*)(pB[g] + (kt + 1) * 512);
            }
            AFrag a = load_afrag(&hbuf[cur][rowA * 512 + kt * 32 + hi * 8]);
#pragma unroll
            for (int g = 0; g < 4; ++g)
                acc[g] = __builtin_amdgcn_wmma_f32_16x16x32_bf16(
                    false, a.v, false, bb[cb][g], (short)0, acc[g], false, false);
        }

        const int nxt = cur ^ 1;
#pragma unroll
        for (int j = 0; j < 8; ++j) {
            const int m = j + 8 * hi;
            float iv = sigmoidf_(acc[0][j]);
            float fv = sigmoidf_(acc[1][j]);
            float gv = tanhf(acc[2][j]);
            float ov = sigmoidf_(acc[3][j]);
            float c  = fv * cst[j] + iv * gv;
            cst[j]   = c;
            unsigned short hb = f2bf(ov * tanhf(c));
            hbuf[nxt][m * 512 + hcol] = hb;
            hcat[(size_t)((b0 + m) * T_ + t) * 1024 + dir * 512 + hcol] = hb;
        }
        __syncthreads();
        cur = nxt;
    }
}

// ---------------------------------------------------------------------------
// emission = exp(hcat @ W_em.T + b_em): (BT x 1024) * (1024 x 128).
// A tile (16x1024) staged in LDS; B fragment double-buffered over K.
// ---------------------------------------------------------------------------
__global__ __launch_bounds__(256, 1) void crf_emission_kernel(
    const unsigned short* __restrict__ hcat,
    const unsigned short* __restrict__ pwem,
    const float* __restrict__ bem,
    float* __restrict__ em) {
    __shared__ unsigned short sA[16 * 1024];   // 32 KB
    const int lane = threadIdx.x & 31;
    const int wave = threadIdx.x >> 5;   // ntile 0..7
    const int rt   = blockIdx.x;
    const int hi   = lane >> 4;
    const int col  = wave * 16 + (lane & 15);

    for (int i = threadIdx.x * 8; i < 16 * 1024; i += 256 * 8) {
        *(ushort8*)&sA[i] =
            *(const ushort8*)&hcat[(size_t)(rt * 16 + (i >> 10)) * 1024 + (i & 1023)];
    }
    __syncthreads();

    v8f acc;
    const float bv = bem[col];
#pragma unroll
    for (int j = 0; j < 8; ++j) acc[j] = bv;

    const unsigned short* pB = pwem + ((size_t)(wave * 32) * 32 + lane) * 16;
    const unsigned short* arow = &sA[(lane & 15) * 1024 + hi * 8];

    v16bf bb[2];
    bb[0] = *(const v16bf*)(pB);
    for (int kt = 0; kt < 32; ++kt) {
        const int cb = kt & 1, nb = cb ^ 1;
        if (kt + 1 < 32) bb[nb] = *(const v16bf*)(pB + (kt + 1) * 512);
        AFrag a = load_afrag(arow + kt * 32);
        acc = __builtin_amdgcn_wmma_f32_16x16x32_bf16(
            false, a.v, false, bb[cb], (short)0, acc, false, false);
    }
#pragma unroll
    for (int j = 0; j < 8; ++j) {
        const int m = rt * 16 + j + 8 * hi;
        em[(size_t)m * L_ + col] = __expf(acc[j]);
    }
}

// Gold-path score: real[b] = sc[0] + sum_{1<=t<len} (sc[t] + exp(trans[lab[t-1],lab[t]]))
__global__ __launch_bounds__(256, 1) void crf_real_kernel(
    const float* __restrict__ em, const int* __restrict__ labels,
    const int* __restrict__ length, const float* __restrict__ trans,
    float* __restrict__ realv) {
    __shared__ float red[256];
    const int b = blockIdx.x;
    const int len = length[b];
    float p = 0.0f;
    for (int t = threadIdx.x; t < T_; t += 256) {
        int lab = labels[b * T_ + t];
        float sc = em[(size_t)(b * T_ + t) * L_ + lab];
        if (t == 0) p += sc;
        else if (t < len) {
            int labp = labels[b * T_ + t - 1];
            p += sc + __expf(trans[labp * L_ + lab]);
        }
    }
    red[threadIdx.x] = p;
    __syncthreads();
    for (int s = 128; s > 0; s >>= 1) {
        if (threadIdx.x < s) red[threadIdx.x] += red[threadIdx.x + s];
        __syncthreads();
    }
    if (threadIdx.x == 0) realv[b] = red[0];
}

// CRF forward scan: one workgroup per batch element; alpha[128] in LDS,
// exp(transition) L2-resident. logZ[b] = logsumexp(alpha_final).
__global__ __launch_bounds__(128, 1) void crf_alpha_kernel(
    const float* __restrict__ em, const int* __restrict__ length,
    const float* __restrict__ texp, float* __restrict__ logZ) {
    __shared__ float sa[L_];
    __shared__ float red[L_];
    const int b = blockIdx.x;
    const int j = threadIdx.x;
    const int len = length[b];
    sa[j] = em[(size_t)(b * T_) * L_ + j];
    __syncthreads();
    for (int t = 1; t < T_; ++t) {
        float nv;
        if (t < len) {
            float mx = -3.4e38f;
#pragma unroll 8
            for (int i = 0; i < L_; ++i)
                mx = fmaxf(mx, sa[i] + texp[i * L_ + j]);
            float sum = 0.0f;
#pragma unroll 8
            for (int i = 0; i < L_; ++i)
                sum += __expf(sa[i] + texp[i * L_ + j] - mx);
            nv = em[(size_t)(b * T_ + t) * L_ + j] + mx + __logf(sum);
        } else {
            nv = sa[j];
        }
        __syncthreads();
        sa[j] = nv;
        __syncthreads();
    }
    red[j] = sa[j];
    __syncthreads();
    for (int s = 64; s > 0; s >>= 1) {
        if (j < s) red[j] = fmaxf(red[j], red[j + s]);
        __syncthreads();
    }
    float mx = red[0];
    __syncthreads();
    red[j] = __expf(sa[j] - mx);
    __syncthreads();
    for (int s = 64; s > 0; s >>= 1) {
        if (j < s) red[j] += red[j + s];
        __syncthreads();
    }
    if (j == 0) logZ[b] = mx + __logf(red[0]);
}

__global__ void crf_final_kernel(const float* __restrict__ logZ,
                                 const float* __restrict__ realv,
                                 float* __restrict__ out) {
    __shared__ float red[64];
    int j = threadIdx.x;
    red[j] = logZ[j] - realv[j];
    __syncthreads();
    for (int s = 32; s > 0; s >>= 1) {
        if (j < s) red[j] += red[j + s];
        __syncthreads();
    }
    if (j == 0) out[0] = red[0];
}

// ---------------------------------------------------------------------------
extern "C" void kernel_launch(void* const* d_in, const int* in_sizes, int n_in,
                              void* d_out, int out_size, void* d_ws, size_t ws_size,
                              hipStream_t stream) {
    (void)in_sizes; (void)n_in; (void)out_size; (void)ws_size;
    const int*   tokens     = (const int*)d_in[0];
    const int*   length     = (const int*)d_in[1];
    const int*   labels     = (const int*)d_in[2];
    const float* emb        = (const float*)d_in[3];
    const float* w_ih_f     = (const float*)d_in[4];
    const float* w_hh_f     = (const float*)d_in[5];
    const float* b_f        = (const float*)d_in[6];
    const float* w_ih_b     = (const float*)d_in[7];
    const float* w_hh_b     = (const float*)d_in[8];
    const float* b_b        = (const float*)d_in[9];
    const float* W_em       = (const float*)d_in[10];
    const float* b_em       = (const float*)d_in[11];
    const float* transition = (const float*)d_in[12];
    float* out = (float*)d_out;

    char* ws = (char*)d_ws;
    size_t off = 0;
    auto walloc = [&](size_t bytes) -> void* {
        void* p = ws + off;
        off = (off + bytes + 255) & ~(size_t)255;
        return p;
    };
    unsigned short* xbf   = (unsigned short*)walloc((size_t)BT_ * E_ * 2);   // 32 MB
    unsigned short* pwihf = (unsigned short*)walloc((size_t)G_ * E_ * 2);
    unsigned short* pwihb = (unsigned short*)walloc((size_t)G_ * E_ * 2);
    unsigned short* pwhhf = (unsigned short*)walloc((size_t)G_ * H_ * 2);
    unsigned short* pwhhb = (unsigned short*)walloc((size_t)G_ * H_ * 2);
    unsigned short* pwem  = (unsigned short*)walloc((size_t)L_ * 1024 * 2);
    float* gxf  = (float*)walloc((size_t)BT_ * G_ * 4);                      // 256 MB
    float* gxb  = (float*)walloc((size_t)BT_ * G_ * 4);                      // 256 MB
    unsigned short* hcat = (unsigned short*)walloc((size_t)BT_ * 1024 * 2);  // 64 MB
    float* em   = (float*)walloc((size_t)BT_ * L_ * 4);                      // 16 MB
    float* texp = (float*)walloc((size_t)L_ * L_ * 4);
    float* logZ = (float*)walloc(64 * 4);
    float* realv = (float*)walloc(64 * 4);

    crf_transexp_kernel<<<(L_ * L_ + 255) / 256, 256, 0, stream>>>(transition, texp);
    crf_embed_kernel<<<(BT_ * E_) / 256, 256, 0, stream>>>(tokens, emb, xbf);
    crf_pack_b_kernel<<<(G_ * E_) / 256, 256, 0, stream>>>(w_ih_f, pwihf, G_, E_);
    crf_pack_b_kernel<<<(G_ * E_) / 256, 256, 0, stream>>>(w_ih_b, pwihb, G_, E_);
    crf_pack_b_kernel<<<(G_ * H_) / 256, 256, 0, stream>>>(w_hh_f, pwhhf, G_, H_);
    crf_pack_b_kernel<<<(G_ * H_) / 256, 256, 0, stream>>>(w_hh_b, pwhhb, G_, H_);
    crf_pack_b_kernel<<<(L_ * 1024) / 256, 256, 0, stream>>>(W_em, pwem, L_, 1024);

    dim3 ggx(BT_ / 16, 4);
    crf_gemm_gx_kernel<<<ggx, 256, 0, stream>>>(xbf, pwihf, b_f, gxf);
    crf_gemm_gx_kernel<<<ggx, 256, 0, stream>>>(xbf, pwihb, b_b, gxb);

    crf_lstm_kernel<<<8, 1024, 0, stream>>>(gxf, gxb, pwhhf, pwhhb, hcat);

    crf_emission_kernel<<<BT_ / 16, 256, 0, stream>>>(hcat, pwem, b_em, em);
    crf_real_kernel<<<B_, 256, 0, stream>>>(em, labels, length, transition, realv);
    crf_alpha_kernel<<<B_, 128, 0, stream>>>(em, length, texp, logZ);
    crf_final_kernel<<<1, 64, 0, stream>>>(logZ, realv, out);
}
